// LstmModel_42030549958715
// MI455X (gfx1250) — matrix operand order repaired and checked by reading, not compiled, not run on hardware
//
#include <hip/hip_runtime.h>

typedef _Float16 v16h __attribute__((ext_vector_type(16)));
typedef float    v8f  __attribute__((ext_vector_type(8)));
typedef unsigned v8u  __attribute__((ext_vector_type(8)));

static __device__ __forceinline__ float fast_tanh(float x) {
#if __has_builtin(__builtin_amdgcn_tanhf)
    return __builtin_amdgcn_tanhf(x);
#else
    // tanh(x) = 2/(1+2^(-2*log2e*x)) - 1
    float e = __builtin_amdgcn_exp2f(-2.885390081777927f * x);
    return fmaf(2.0f, __builtin_amdgcn_rcpf(1.0f + e), -1.0f);
#endif
}

// One wave = 16 batch elements. Tiles pair the SAME gate of both LSTMs:
//   tile0=i1|i2, tile1=f1|f2, tile2=g1|g2, tile3=o1|o2
// so D tile r: lanes0-15 = gate_r(LSTM1)[batch], lanes16-31 = gate_r(LSTM2)[batch]
// and the whole c/h update is full-wave dense.
// K layout of the shared B operand: K0-7=2*h1, K8-15=2*h2, K16=x1, K17=x2, K18=1.0
// (W_ih and fused bias live in A at K16-18, so the WMMA C operand is zero).
// Folded scalings:
//   * sigmoid tiles (i,f,o) pre-scaled by 0.5 in A -> sigmoid = tanh(D)*0.5+0.5
//   * hidden state tracked as 2h = fma(tanh_o, tc, tc); all h-columns of A
//     (and of the MLP weights) carry an extra 0.5.
__global__ __launch_bounds__(256)
void lstm2_wmma_kernel(const float* __restrict__ x,
                       const float* __restrict__ bvec,
                       const float* __restrict__ Wih1, const float* __restrict__ Whh1,
                       const float* __restrict__ bih1, const float* __restrict__ bhh1,
                       const float* __restrict__ Wih2, const float* __restrict__ Whh2,
                       const float* __restrict__ bih2, const float* __restrict__ bhh2,
                       const float* __restrict__ Wfc1, const float* __restrict__ bfc1,
                       const float* __restrict__ Wfc2, const float* __restrict__ bfc2,
                       float* __restrict__ out, int T)
{
    const int  lane = threadIdx.x & 31;
    const int  wave = threadIdx.x >> 5;
    const int  n    = lane & 15;        // batch column; also A-matrix row M
    const bool hi   = lane >= 16;
    const int  batchBase = (blockIdx.x * 8 + wave) * 16;

    // ---- A tiles. Row M<8 -> LSTM1 gate (goff+M); M>=8 -> LSTM2 gate (goff+M-8).
    // f16 A layout: lanes0-15 elems0-7=K0-7, elems8-15=K16-23;
    //               lanes16-31 elems0-7=K8-15, elems8-15=K24-31.
    const bool  rowL2  = n >= 8;
    const int   g8     = n & 7;
    const float whmask = (hi == rowL2) ? 1.f : 0.f;  // Whh1 rows live lo, Whh2 rows live hi
    const float lomask = hi ? 0.f : 1.f;             // x/bias slots are K16-18 (lo lanes only)
    const float* WhhX = rowL2 ? Whh2 : Whh1;
    const float* WihX = rowL2 ? Wih2 : Wih1;
    const float* biX  = rowL2 ? bih2 : bih1;
    const float* bhX  = rowL2 ? bhh2 : bhh1;

    v16h A[4];
#pragma unroll
    for (int t = 0; t < 4; ++t) {                    // goff = 0(i), 8(f), 16(g), 24(o)
        const int   g      = t * 8 + g8;
        const float tscale = (t == 2) ? 1.0f : 0.5f; // fold sigmoid's 0.5 pre-scale
        const float* wr = WhhX + g * 8;
        v16h a;
#pragma unroll
        for (int e = 0; e < 8; ++e)                  // extra 0.5: B carries 2h
            a[e] = (_Float16)(wr[e] * whmask * tscale * 0.5f);
        const float wih = WihX[g] * tscale;
        const float bia = (biX[g] + bhX[g]) * tscale;
        a[8]  = (_Float16)((rowL2 ? 0.f : wih) * lomask);  // K16 = x1 slot (LSTM1 rows)
        a[9]  = (_Float16)((rowL2 ? wih : 0.f) * lomask);  // K17 = x2 slot (LSTM2 rows)
        a[10] = (_Float16)(bia * lomask);                  // K18 = 1.0 slot (fused bias)
#pragma unroll
        for (int e = 11; e < 16; ++e) a[e] = (_Float16)0.f;
        A[t] = a;
    }

    v8f Cz;
#pragma unroll
    for (int r = 0; r < 8; ++r) Cz[r] = 0.f;

    // ---- state: cc full-wave (lo=c1, hi=c2); hw = lo-lane B words (2h1 pairs, 2h2 pairs)
    float    cc[8];
    unsigned hw[8];
#pragma unroll
    for (int r = 0; r < 8; ++r) { cc[r] = 0.f; hw[r] = 0u; }

    // x row; both lane halves load the same (cached) float4 per 2 steps.
    const float4* xrow = reinterpret_cast<const float4*>(x) +
                         (size_t)(batchBase + n) * (size_t)(T >> 1);
    float4 xb0 = xrow[0];
    float4 xb1 = xrow[1];

    for (int t0 = 0; t0 < T; t0 += 4) {
        const int nxt = (t0 + 4 < T) ? ((t0 + 4) >> 1) : 0;   // uniform, no EXEC change
        const float4 nb0 = xrow[nxt];
        const float4 nb1 = xrow[nxt + 1];
#pragma unroll
        for (int j = 0; j < 4; ++j) {
            const float4 xb = (j & 2) ? xb1 : xb0;
            const float  xa = (j & 1) ? xb.z : xb.x;   // x[...,0] at this t
            const float  xc = (j & 1) ? xb.w : xb.y;   // x[...,1]
            auto pkx = __builtin_amdgcn_cvt_pkrtz(xa, xc);

            // B operand: words0-3 = own 2h pairs (lo: K0-7=2h1, hi: K8-15=2h2); lo words4,5 = [x1,x2 | 1,0].
            // Remaining words are finite residue multiplied by structural zeros in A.
            v8u bu;
#pragma unroll
            for (int w = 0; w < 8; ++w) bu[w] = hw[w];
            bu[4] = hi ? hw[4] : __builtin_bit_cast(unsigned, pkx);
            bu[5] = hi ? hw[5] : 0x00003C00u;
            const v16h Bm = __builtin_bit_cast(v16h, bu);

            v8f D0 = __builtin_amdgcn_wmma_f32_16x16x32_f16(false, A[0], false, Bm, (short)0, Cz, false, false); // 0.5*i
            v8f D1 = __builtin_amdgcn_wmma_f32_16x16x32_f16(false, A[1], false, Bm, (short)0, Cz, false, false); // 0.5*f
            v8f D2 = __builtin_amdgcn_wmma_f32_16x16x32_f16(false, A[2], false, Bm, (short)0, Cz, false, false); // g
            v8f D3 = __builtin_amdgcn_wmma_f32_16x16x32_f16(false, A[3], false, Bm, (short)0, Cz, false, false); // 0.5*o

            float hh[8];
#pragma unroll
            for (int r = 0; r < 8; ++r) {
                const float ti = fast_tanh(D0[r]);
                const float tf = fast_tanh(D1[r]);
                const float gv = fast_tanh(D2[r]);
                const float to = fast_tanh(D3[r]);
                const float iv = fmaf(ti, 0.5f, 0.5f);
                const float fv = fmaf(tf, 0.5f, 0.5f);
                cc[r] = fmaf(fv, cc[r], iv * gv);
                const float tc = fast_tanh(cc[r]);
                hh[r] = fmaf(to, tc, tc);          // = 2*o*tanh(c); lo=2h1, hi=2h2
            }
            // pack first (lo lanes -> h1 pairs, hi lanes -> h2 pairs), then swap halves
#pragma unroll
            for (int p = 0; p < 4; ++p) {
                auto q = __builtin_amdgcn_cvt_pkrtz(hh[2*p], hh[2*p+1]);
                const unsigned pw = __builtin_bit_cast(unsigned, q);
                hw[p]     = pw;                                           // K0-7  = 2h1
                hw[4 + p] = (unsigned)__shfl_xor((int)pw, 16, 32);        // other half's pairs; in B only as residue
            }
        }
        xb0 = nb0; xb1 = nb1;
    }

    // ---- final MLP: hid = relu(Wfc1 @ [h1,h2,b] + bfc1); out = Wfc2 @ hid + bfc2
    // B carries 2h, so the h-columns of Wfc1 get the same 0.5 fold.
    v16h Af;
    {
        const float* wr = Wfc1 + n * 20;
        float wv[20];
#pragma unroll
        for (int k = 0; k < 20; ++k) wv[k] = wr[k];
#pragma unroll
        for (int e = 0; e < 8; ++e)                                         // K0-7 | K8-15
            Af[e] = (_Float16)((hi ? wv[8 + e] : wv[e]) * 0.5f);
#pragma unroll
        for (int e = 0; e < 4; ++e) Af[8 + e] = (_Float16)(hi ? 0.f : wv[16 + e]);  // K16-19 (b)
#pragma unroll
        for (int e = 12; e < 16; ++e) Af[e] = (_Float16)0.f;
    }
    v16h Bf;
    {
        const float4 bb = reinterpret_cast<const float4*>(bvec)[batchBase + n];
        auto q0 = __builtin_amdgcn_cvt_pkrtz(bb.x, bb.y);
        auto q1 = __builtin_amdgcn_cvt_pkrtz(bb.z, bb.w);
        v8u t;
#pragma unroll
        for (int w = 0; w < 8; ++w) t[w] = hw[w];               // residue x A-zeros
        t[4] = hi ? hw[4] : __builtin_bit_cast(unsigned, q0);   // K16,17 = b0,b1
        t[5] = hi ? hw[5] : __builtin_bit_cast(unsigned, q1);   // K18,19 = b2,b3
        Bf = __builtin_bit_cast(v16h, t);
    }
    v8f Cf;
#pragma unroll
    for (int r = 0; r < 8; ++r) Cf[r] = bfc1[r + (hi ? 8 : 0)];
    v8f Df = __builtin_amdgcn_wmma_f32_16x16x32_f16(false, Af, false, Bf, (short)0, Cf, false, false);

    float acc = 0.f;
#pragma unroll
    for (int r = 0; r < 8; ++r) {
        const float hd = fmaxf(Df[r], 0.f);
        const float w2 = hi ? Wfc2[r + 8] : Wfc2[r];
        acc = fmaf(hd, w2, acc);
    }
    acc += __shfl_xor(acc, 16, 32);
    if (!hi) out[batchBase + n] = acc + bfc2[0];
}

extern "C" void kernel_launch(void* const* d_in, const int* in_sizes, int n_in,
                              void* d_out, int out_size, void* d_ws, size_t ws_size,
                              hipStream_t stream) {
    (void)n_in; (void)out_size; (void)d_ws; (void)ws_size;
    const float* x    = (const float*)d_in[0];
    const float* bvec = (const float*)d_in[1];
    const float* Wih1 = (const float*)d_in[2];
    const float* Whh1 = (const float*)d_in[3];
    const float* bih1 = (const float*)d_in[4];
    const float* bhh1 = (const float*)d_in[5];
    const float* Wih2 = (const float*)d_in[6];
    const float* Whh2 = (const float*)d_in[7];
    const float* bih2 = (const float*)d_in[8];
    const float* bhh2 = (const float*)d_in[9];
    const float* Wfc1 = (const float*)d_in[10];
    const float* bfc1 = (const float*)d_in[11];
    const float* Wfc2 = (const float*)d_in[12];
    const float* bfc2 = (const float*)d_in[13];

    const int B = in_sizes[1] / 4;          // b is [B,4]
    const int T = in_sizes[0] / (2 * B);    // x is [B,T,2]
    const int blocks = B / 128;             // 8 waves/block * 16 batch/wave

    lstm2_wmma_kernel<<<blocks, 256, 0, stream>>>(
        x, bvec, Wih1, Whh1, bih1, bhh1, Wih2, Whh2, bih2, bhh2,
        Wfc1, bfc1, Wfc2, bfc2, (float*)d_out, T);
}
